// KNNAttention_88545045774776
// MI455X (gfx1250) — matrix operand order repaired and checked by reading, not compiled, not run on hardware
//
#include <hip/hip_runtime.h>
#include <hip/hip_bf16.h>

// ---------------------------------------------------------------------------
// Problem constants (reference: B=2, N=2048, DIM=1024, H=8, DH=64, K=32)
// ---------------------------------------------------------------------------
#define PB    2
#define PN    2048
#define PDIM  1024
#define PH    8
#define PDH   64
#define PK    32
#define PINNER (PH * PDH)     // 512
#define PKV    (2 * PDH)      // 128
#define NEG_BIG (-3.0e38f)

typedef __attribute__((ext_vector_type(16))) __bf16       v16bf;
typedef __attribute__((ext_vector_type(8)))  float        v8f;
typedef __attribute__((ext_vector_type(8)))  unsigned int v8u;

// fp32 -> bf16 (round-to-nearest-even)
__device__ __forceinline__ unsigned short f2bf(float a) {
  unsigned int ua = __builtin_bit_cast(unsigned int, a);
  ua += 0x7FFFu + ((ua >> 16) & 1u);
  return (unsigned short)(ua >> 16);
}
__device__ __forceinline__ unsigned int pack_bf16(float a, float b) {
  return (unsigned int)f2bf(a) | ((unsigned int)f2bf(b) << 16);
}
__device__ __forceinline__ float bf2f(unsigned short u) {
  return __builtin_bit_cast(float, ((unsigned int)u) << 16);
}

// A-matrix (16x32, 16-bit) per-lane K index (pair base) for VGPR v, lane-half.
__device__ __forceinline__ int a_kidx(int v, int half) {
  return half * 16 + v * 2;
}

// ---------------------------------------------------------------------------
// One-shot conversion kernels (streaming; tiny vs. the 536MB mem_kv stream)
// ---------------------------------------------------------------------------
// Row-major f32 -> row-major bf16 (vectorized: float4 -> 2 dwords)
__global__ void cvt_bf16_rm(const float* __restrict__ src,
                            unsigned short* __restrict__ dst, size_t n4) {
  size_t i = (size_t)blockIdx.x * blockDim.x + threadIdx.x;
  if (i >= n4) return;
  float4 f = ((const float4*)src)[i];
  unsigned int lo = pack_bf16(f.x, f.y);
  unsigned int hi = pack_bf16(f.z, f.w);
  ((uint2*)dst)[i] = make_uint2(lo, hi);
}

// W[Kd][Ncols] f32 -> Wt[Ncols][Kd] bf16 (coalesced writes along k)
__global__ void cvt_bf16_tr(const float* __restrict__ W,
                            unsigned short* __restrict__ Wt, int Kd, int Ncols) {
  size_t t = (size_t)blockIdx.x * blockDim.x + threadIdx.x;
  size_t total = (size_t)Kd * Ncols;
  if (t >= total) return;
  int col = (int)(t / Kd);
  int k   = (int)(t % Kd);
  Wt[(size_t)col * Kd + k] = f2bf(W[(size_t)k * Ncols + col]);
}

// ---------------------------------------------------------------------------
// GEMM: C[M,Ncols] = A[M,Kd](bf16 row-major) @ Wt[Ncols,Kd](bf16, K-contig)
// + optional bias.  Wave -> 16x32 D (2 tiles sharing the A-fragment).
// Block = 128 threads = 4 waves -> 16x128 tile.  grid = (M/16, Ncols/128).
// Inner loop: 24 dword (-> b128-clause) loads + 2 WMMA per 32-k step.
// ---------------------------------------------------------------------------
template <bool OUT_BF16>
__global__ __launch_bounds__(128)
void gemm_bf16_wmma(const unsigned short* __restrict__ A,
                    const unsigned short* __restrict__ Wt,
                    const float* __restrict__ bias, void* __restrict__ Cout,
                    int M, int Kd, int Ncols) {
  const int lane = threadIdx.x & 31;
  const int wave = threadIdx.x >> 5;
  const int half = lane >> 4;
  const int l16  = lane & 15;
  const int row0 = blockIdx.x * 16;
  const int colbase = blockIdx.y * 128 + wave * 32;

  const unsigned int* arow = (const unsigned int*)(A + (size_t)(row0 + l16) * Kd);
  const unsigned int* bcol[2] = {
      (const unsigned int*)(Wt + (size_t)(colbase + l16) * Kd),
      (const unsigned int*)(Wt + (size_t)(colbase + 16 + l16) * Kd)};

  v8f acc[2] = {v8f{}, v8f{}};
  for (int k0 = 0; k0 < Kd; k0 += 32) {
    v8u au;
#pragma unroll
    for (int v = 0; v < 8; ++v) au[v] = arow[(k0 + a_kidx(v, half)) >> 1];
    v16bf a = __builtin_bit_cast(v16bf, au);
#pragma unroll
    for (int nt = 0; nt < 2; ++nt) {
      v8u bu;
#pragma unroll
      for (int j = 0; j < 8; ++j) bu[j] = bcol[nt][(k0 + half * 16 + j * 2) >> 1];
      v16bf b = __builtin_bit_cast(v16bf, bu);
      acc[nt] = __builtin_amdgcn_wmma_f32_16x16x32_bf16(false, a, false, b,
                                                        (short)0, acc[nt], false, false);
    }
  }

  // branch-free epilogue; bias load hoisted (per-lane constant)
  float bv[2] = {0.0f, 0.0f};
  if (bias) { bv[0] = bias[colbase + l16]; bv[1] = bias[colbase + 16 + l16]; }
#pragma unroll
  for (int nt = 0; nt < 2; ++nt) {
    const int col = colbase + nt * 16 + l16;
#pragma unroll
    for (int r = 0; r < 8; ++r) {
      const int row = row0 + half * 8 + r;
      const float o = acc[nt][r] + bv[nt];
      if (OUT_BF16)
        ((unsigned short*)Cout)[(size_t)row * Ncols + col] = f2bf(o);
      else
        ((float*)Cout)[(size_t)row * Ncols + col] = o;
    }
  }
}

// ---------------------------------------------------------------------------
// Fused attention: flash-style causal local attention (WMMA) + memory
// attention over K+1 kv pairs (coalesced single-pass stream w/ prefetch) +
// sigmoid gate.  One wave per 16 query rows of one (b,h).
// grid = (N/16, B*H), block = 32.
// ---------------------------------------------------------------------------
__global__ __launch_bounds__(32)
void fused_attn(const unsigned short* __restrict__ q,   // [B*N, INNER] bf16
                const unsigned short* __restrict__ kvb, // [B*N, 2*DH]  bf16
                const float* __restrict__ mem_kv,       // [B, H*N, K, 2, DH]
                const unsigned char* __restrict__ mem_mask, // [B, H*N, K]
                const float* __restrict__ null_k,
                const float* __restrict__ null_v,
                const float* __restrict__ gate,         // [H]
                unsigned short* __restrict__ comb)      // [B*N, INNER] bf16
{
  __shared__ unsigned int Pls[16 * 16];  // P tile 16x32 bf16, row-major dwords
  __shared__ float Ols[16][PDH];         // normalized local values

  const int lane = threadIdx.x & 31;
  const int half = lane >> 4;
  const int l16  = lane & 15;
  const int ibase = blockIdx.x * 16;
  const int b = blockIdx.y / PH;
  const int h = blockIdx.y % PH;
  const float scale = 0.125f;  // DH^-0.5

  // ---- Q fragments: two 16x32 bf16 A-frags (aligned dword loads) ----
  v16bf qa[2];
  {
    const unsigned int* qr =
        (const unsigned int*)(q + ((size_t)b * PN + ibase + l16) * PINNER + h * PDH);
    v8u qu0, qu1;
#pragma unroll
    for (int v = 0; v < 8; ++v) {
      const int ka = a_kidx(v, half);
      qu0[v] = qr[ka >> 1];
      qu1[v] = qr[(32 + ka) >> 1];
    }
    qa[0] = __builtin_bit_cast(v16bf, qu0);
    qa[1] = __builtin_bit_cast(v16bf, qu1);
  }

  float mrow[8], lrow[8];
  v8f O[4] = {};
#pragma unroll
  for (int r = 0; r < 8; ++r) { mrow[r] = NEG_BIG; lrow[r] = 0.0f; }

  // ---- flash loop over 32-key tiles (causal bound) ----
  for (int jb = 0; jb < ibase + 16; jb += 32) {
    v8f S[2] = {v8f{}, v8f{}};
#pragma unroll
    for (int nt = 0; nt < 2; ++nt) {
      const unsigned int* krow =
          (const unsigned int*)(kvb + ((size_t)b * PN + jb + nt * 16 + l16) * PKV);
#pragma unroll
      for (int s = 0; s < 2; ++s) {
        v8u bu;
#pragma unroll
        for (int j = 0; j < 8; ++j) bu[j] = krow[(s * 32 + half * 16 + j * 2) >> 1];
        v16bf bb = __builtin_bit_cast(v16bf, bu);
        S[nt] = __builtin_amdgcn_wmma_f32_16x16x32_bf16(false, qa[s], false, bb,
                                                        (short)0, S[nt], false, false);
      }
    }
    // scale + causal mask + row stats (16-lane halves hold the 16 columns)
    float tmax[8], tsum[8];
#pragma unroll
    for (int r = 0; r < 8; ++r) {
      const int row = ibase + half * 8 + r;
#pragma unroll
      for (int nt = 0; nt < 2; ++nt) {
        const int col = jb + nt * 16 + l16;
        float v = S[nt][r] * scale;
        if (col > row) v = NEG_BIG;
        S[nt][r] = v;
      }
      tmax[r] = fmaxf(S[0][r], S[1][r]);
    }
#pragma unroll
    for (int m = 1; m < 16; m <<= 1)
#pragma unroll
      for (int r = 0; r < 8; ++r)
        tmax[r] = fmaxf(tmax[r], __shfl_xor(tmax[r], m, 32));

#pragma unroll
    for (int r = 0; r < 8; ++r) {
      const float nm = fmaxf(mrow[r], tmax[r]);
      const float cs = __expf(mrow[r] - nm);
      mrow[r] = nm;
      tsum[r] = 0.0f;
#pragma unroll
      for (int nt = 0; nt < 2; ++nt) {
        const float p = __expf(S[nt][r] - nm);
        S[nt][r] = p;
        tsum[r] += p;
      }
      lrow[r] *= cs;
#pragma unroll
      for (int dt = 0; dt < 4; ++dt) O[dt][r] *= cs;
    }
#pragma unroll
    for (int m = 1; m < 16; m <<= 1)
#pragma unroll
      for (int r = 0; r < 8; ++r)
        tsum[r] += __shfl_xor(tsum[r], m, 32);
#pragma unroll
    for (int r = 0; r < 8; ++r) lrow[r] += tsum[r];

    // P (D-layout) -> LDS row-major bf16 -> reload as A-frag (16x32)
    __syncthreads();
    {
      unsigned short* Pls16 = (unsigned short*)Pls;
#pragma unroll
      for (int r = 0; r < 8; ++r) {
        const int m = half * 8 + r;
        Pls16[m * 32 + l16]      = f2bf(S[0][r]);
        Pls16[m * 32 + 16 + l16] = f2bf(S[1][r]);
      }
    }
    __syncthreads();
    v8u pu;
#pragma unroll
    for (int v = 0; v < 8; ++v) pu[v] = Pls[l16 * 16 + (a_kidx(v, half) >> 1)];
    v16bf pa = __builtin_bit_cast(v16bf, pu);

    // O += P @ V   (V: 32 keys x 64 dims, 4 column tiles)
#pragma unroll
    for (int dt = 0; dt < 4; ++dt) {
      v8u vu;
#pragma unroll
      for (int j = 0; j < 8; ++j) {
        const int kk = half * 16 + j * 2;  // key offset in tile
        const unsigned short* vp =
            kvb + ((size_t)b * PN + jb + kk) * PKV + PDH + dt * 16 + l16;
        const unsigned int v0 = vp[0];
        const unsigned int v1 = vp[PKV];
        vu[j] = v0 | (v1 << 16);
      }
      v16bf vb = __builtin_bit_cast(v16bf, vu);
      O[dt] = __builtin_amdgcn_wmma_f32_16x16x32_bf16(false, pa, false, vb,
                                                      (short)0, O[dt], false, false);
    }
  }

  // normalize local values into LDS for the combine phase
  __syncthreads();
#pragma unroll
  for (int r = 0; r < 8; ++r) {
    const float inv = 1.0f / lrow[r];
#pragma unroll
    for (int dt = 0; dt < 4; ++dt)
      Ols[half * 8 + r][dt * 16 + l16] = O[dt][r] * inv;
  }
  __syncthreads();

  // ---- memory attention + gate (streams mem_kv exactly once, coalesced) ----
  const float g = 1.0f / (1.0f + __expf(-gate[h]));
  for (int i = 0; i < 16; ++i) {
    const int row = ibase + i;
    const size_t hn = (size_t)b * (PH * PN) + (size_t)h * PN + row;
    const unsigned short* qr = q + ((size_t)b * PN + row) * PINNER + h * PDH;
    const float q0 = bf2f(qr[lane]);
    const float q1 = bf2f(qr[lane + 32]);

    // null-key score
    float sn = q0 * null_k[lane] + q1 * null_k[lane + 32];
#pragma unroll
    for (int m = 1; m < 32; m <<= 1) sn += __shfl_xor(sn, m, 32);
    sn *= scale;

    // lane j scores memory key j (each lane streams one 256B key row)
    const float* mk = mem_kv + (hn * PK + lane) * (2 * PDH);
    __builtin_prefetch(mk + 2 * PDH * PK, 0, 1);  // next row's keys
    float sj = 0.0f;
#pragma unroll
    for (int d4 = 0; d4 < 16; ++d4) {
      const float4 kk4 = ((const float4*)mk)[d4];
      const int d = d4 * 4;
      const float src = (d < 32) ? q0 : q1;
      sj += kk4.x * __shfl(src, (d + 0) & 31, 32);
      sj += kk4.y * __shfl(src, (d + 1) & 31, 32);
      sj += kk4.z * __shfl(src, (d + 2) & 31, 32);
      sj += kk4.w * __shfl(src, (d + 3) & 31, 32);
    }
    sj *= scale;
    if (!mem_mask[hn * PK + lane]) sj = NEG_BIG;

    // softmax over {null} U 32 keys
    float mx = sj;
#pragma unroll
    for (int m = 1; m < 32; m <<= 1) mx = fmaxf(mx, __shfl_xor(mx, m, 32));
    mx = fmaxf(mx, sn);
    const float pj = __expf(sj - mx);
    const float pn = __expf(sn - mx);
    float lsum = pj;
#pragma unroll
    for (int m = 1; m < 32; m <<= 1) lsum += __shfl_xor(lsum, m, 32);
    lsum += pn;

    // weighted value sum: lane owns dims (lane, lane+32); coalesced stream
    float a0 = pn * null_v[lane];
    float a1 = pn * null_v[lane + 32];
    const float* mvb = mem_kv + hn * PK * (2 * PDH) + PDH;
    for (int j = 0; j < PK; ++j) {
      const float p = __shfl(pj, j, 32);
      const float* mv = mvb + (size_t)j * (2 * PDH);
      __builtin_prefetch(mv + 2 * (2 * PDH), 0, 1);  // 2 rows ahead
      a0 += p * mv[lane];
      a1 += p * mv[lane + 32];
    }
    const float inv = 1.0f / lsum;

    // gate-combine with local values; emit bf16 (A-operand of final GEMM)
    unsigned short* cr = comb + ((size_t)b * PN + row) * PINNER + h * PDH;
    cr[lane]      = f2bf(Ols[i][lane]      * g + (a0 * inv) * (1.0f - g));
    cr[lane + 32] = f2bf(Ols[i][lane + 32] * g + (a1 * inv) * (1.0f - g));
  }
}

// ---------------------------------------------------------------------------
// Host-side launcher
// ---------------------------------------------------------------------------
extern "C" void kernel_launch(void* const* d_in, const int* in_sizes, int n_in,
                              void* d_out, int out_size, void* d_ws, size_t ws_size,
                              hipStream_t stream) {
  const float* x        = (const float*)d_in[0];  // [B,N,DIM]
  const float* Wq       = (const float*)d_in[1];  // [DIM,INNER]
  const float* Wkv      = (const float*)d_in[2];  // [DIM,2*DH]
  const float* Wo       = (const float*)d_in[3];  // [INNER,DIM]
  const float* bo       = (const float*)d_in[4];  // [DIM]
  const float* null_k   = (const float*)d_in[5];  // [DH]
  const float* null_v   = (const float*)d_in[6];  // [DH]
  const float* gate     = (const float*)d_in[7];  // [H,1,1]
  const float* mem_kv   = (const float*)d_in[8];  // [B,H*N,K,2,DH]
  const unsigned char* mem_mask = (const unsigned char*)d_in[9]; // [B,H*N,K]

  float* out = (float*)d_out;                     // [B,N,DIM] f32

  const int M = PB * PN;                          // 4096 rows
  unsigned short* xb   = (unsigned short*)d_ws;               // [M, DIM]
  unsigned short* wq_t = xb   + (size_t)M * PDIM;             // [INNER, DIM]
  unsigned short* wkv_t= wq_t + (size_t)PINNER * PDIM;        // [2DH, DIM]
  unsigned short* wo_t = wkv_t+ (size_t)PKV * PDIM;           // [DIM, INNER]
  unsigned short* qb   = wo_t + (size_t)PDIM * PINNER;        // [M, INNER]
  unsigned short* kvb  = qb   + (size_t)M * PINNER;           // [M, 2DH]
  unsigned short* comb = kvb  + (size_t)M * PKV;              // [M, INNER]

  // --- one-shot bf16 conversion / weight transposition ---
  {
    size_t n4 = (size_t)M * PDIM / 4;
    cvt_bf16_rm<<<dim3((unsigned)((n4 + 255) / 256)), 256, 0, stream>>>(x, xb, n4);
    size_t tq = (size_t)PDIM * PINNER;
    cvt_bf16_tr<<<dim3((unsigned)((tq + 255) / 256)), 256, 0, stream>>>(Wq, wq_t, PDIM, PINNER);
    size_t tk = (size_t)PDIM * PKV;
    cvt_bf16_tr<<<dim3((unsigned)((tk + 255) / 256)), 256, 0, stream>>>(Wkv, wkv_t, PDIM, PKV);
    size_t to = (size_t)PINNER * PDIM;
    cvt_bf16_tr<<<dim3((unsigned)((to + 255) / 256)), 256, 0, stream>>>(Wo, wo_t, PINNER, PDIM);
  }

  dim3 gblk(128);
  // q = x @ Wq   (bf16 out)
  gemm_bf16_wmma<true><<<dim3(M / 16, PINNER / 128), gblk, 0, stream>>>(
      xb, wq_t, nullptr, qb, M, PDIM, PINNER);
  // kv = x @ Wkv (bf16 out)
  gemm_bf16_wmma<true><<<dim3(M / 16, PKV / 128), gblk, 0, stream>>>(
      xb, wkv_t, nullptr, kvb, M, PDIM, PKV);
  // fused local flash attention + memory attention + gate (bf16 out)
  fused_attn<<<dim3(PN / 16, PB * PH), dim3(32), 0, stream>>>(
      qb, kvb, mem_kv, mem_mask, null_k, null_v, gate, comb);
  // out = comb @ Wo + bo  (f32 out)
  gemm_bf16_wmma<false><<<dim3(M / 16, PDIM / 128), gblk, 0, stream>>>(
      comb, wo_t, bo, out, M, PINNER, PDIM);
}
